// Switch_33578054320454
// MI455X (gfx1250) — matrix operand order repaired and checked by reading, not compiled, not run on hardware
//
#include <hip/hip_runtime.h>
#include <hip/hip_bf16.h>

// ---- problem constants (match reference) ----
constexpr int kD = 1024;   // embed dim == ff hidden
constexpr int kE = 16;     // experts
constexpr int kT = 8192;   // tokens
constexpr int kC = 512;    // expert capacity

typedef __attribute__((ext_vector_type(16))) __bf16 bf16x16;
typedef __attribute__((ext_vector_type(8)))  __bf16 bf16x8;
typedef __attribute__((ext_vector_type(8)))  float  f32x8;

// -------------------------------------------------------------------------
// 1) Weight convert + transpose: W[e][d][f] (f32) -> Wt[e][f][d] (bf16)
// -------------------------------------------------------------------------
__global__ __launch_bounds__(256)
void convert_w_kernel(const float* __restrict__ W1, const float* __restrict__ W2,
                      __bf16* __restrict__ W1t, __bf16* __restrict__ W2t) {
    __shared__ float tile[32][33];
    int z = blockIdx.z;
    const float* W  = (z < kE) ? W1  : W2;
    __bf16*      Wt = (z < kE) ? W1t : W2t;
    int e = z & (kE - 1);
    const float* Wm = W  + (size_t)e * kD * kD;
    __bf16*      Wo = Wt + (size_t)e * kD * kD;
    int f0 = blockIdx.x * 32;
    int d0 = blockIdx.y * 32;
    int tx = threadIdx.x & 31, ty = threadIdx.x >> 5;
#pragma unroll
    for (int r = 0; r < 32; r += 8)
        tile[ty + r][tx] = Wm[(size_t)(d0 + ty + r) * kD + f0 + tx];
    __syncthreads();
#pragma unroll
    for (int r = 0; r < 32; r += 8)
        Wo[(size_t)(f0 + ty + r) * kD + d0 + tx] = (__bf16)tile[tx][ty + r];
}

// -------------------------------------------------------------------------
// 2) Router
// -------------------------------------------------------------------------
__global__ __launch_bounds__(256)
void router_kernel(const float* __restrict__ x, const float* __restrict__ Wr,
                   const float* __restrict__ br, int* __restrict__ idxArr,
                   float* __restrict__ gateArr) {
    __shared__ float xs[kD];
    __shared__ float partial[kE][17];
    __shared__ float logits[kE];
    int t = blockIdx.x;
    const float* xr = x + (size_t)t * kD;
    for (int i = threadIdx.x; i < kD; i += 256) xs[i] = xr[i];
    __syncthreads();
    int e  = threadIdx.x & 15;
    int dp = threadIdx.x >> 4;
    float acc = 0.f;
#pragma unroll 4
    for (int j = 0; j < kD / 16; ++j) {
        int d = dp * (kD / 16) + j;
        acc += xs[d] * Wr[(size_t)d * kE + e];
    }
    partial[e][dp] = acc;
    __syncthreads();
    if (threadIdx.x < kE) {
        float s = br[threadIdx.x];
#pragma unroll
        for (int p = 0; p < 16; ++p) s += partial[threadIdx.x][p];
        logits[threadIdx.x] = s;
    }
    __syncthreads();
    if (threadIdx.x == 0) {
        float m = logits[0]; int am = 0;
#pragma unroll
        for (int i = 1; i < kE; ++i) { if (logits[i] > m) { m = logits[i]; am = i; } }
        float s = 0.f;
#pragma unroll
        for (int i = 0; i < kE; ++i) s += __expf(logits[i] - m);
        idxArr[t]  = am;
        gateArr[t] = 1.0f / s;
    }
}

// -------------------------------------------------------------------------
// 3) Ordered slot assignment (deterministic ballot-based scan)
// -------------------------------------------------------------------------
__global__ __launch_bounds__(256)
void assign_kernel(const int* __restrict__ idxArr, const float* __restrict__ gateArr,
                   int* __restrict__ slot_token, int* __restrict__ slotArr,
                   float* __restrict__ wcomb) {
    __shared__ int waveCnt[8][kE];
    __shared__ int waveOff[8][kE];
    __shared__ int baseCnt[kE];
    int tid = threadIdx.x;
    for (int i = tid; i < kE * kC; i += 256) slot_token[i] = -1;
    if (tid < kE) baseCnt[tid] = 0;
    __syncthreads();
    int lane = tid & 31, wv = tid >> 5;
    unsigned ltmask = (1u << lane) - 1u;
    for (int t0 = 0; t0 < kT; t0 += 256) {
        int t  = t0 + tid;
        int ex = idxArr[t];
        int rank = 0;
#pragma unroll
        for (int e = 0; e < kE; ++e) {
            unsigned bal = (unsigned)__ballot(ex == e);
            if (lane == 0) waveCnt[wv][e] = __popc(bal);
            if (ex == e)   rank = __popc(bal & ltmask);
        }
        __syncthreads();
        if (tid == 0) {
            for (int e = 0; e < kE; ++e) {
                int run = baseCnt[e];
                for (int w = 0; w < 8; ++w) { waveOff[w][e] = run; run += waveCnt[w][e]; }
                baseCnt[e] = run;
            }
        }
        __syncthreads();
        int pos = waveOff[wv][ex] + rank + 1;     // 1-based (slot 0 unused)
        bool kept = pos < kC;
        wcomb[t]   = kept ? gateArr[t] : 0.f;
        slotArr[t] = kept ? pos : (kC - 1);
        if (kept) slot_token[ex * kC + pos] = t;
        __syncthreads();
    }
}

// -------------------------------------------------------------------------
// 4) Dispatch: ei[e][slot][:] = w[tok] * x[tok][:]  (bf16)
// -------------------------------------------------------------------------
__global__ __launch_bounds__(256)
void dispatch_kernel(const float* __restrict__ x, const float* __restrict__ wcomb,
                     const int* __restrict__ slot_token, __bf16* __restrict__ ei) {
    int s = blockIdx.x;
    int tok = slot_token[s];
    __bf16* dst = ei + (size_t)s * kD;
    if (tok < 0) {
        for (int i = threadIdx.x; i < kD; i += 256) dst[i] = (__bf16)0.0f;
    } else {
        float w = wcomb[tok];
        const float* src = x + (size_t)tok * kD;
        for (int i = threadIdx.x; i < kD; i += 256) dst[i] = (__bf16)(w * src[i]);
    }
}

// -------------------------------------------------------------------------
// 5) Per-expert bf16 WMMA GEMM, double-buffered LDS fed by
//    GLOBAL_LOAD_ASYNC_TO_LDS_B128 (ASYNCcnt-tracked DMA, overlaps WMMAs).
//    Block tile 128x128, 8 waves (2Mx4N), wave = 64x32 = 4x2 WMMA tiles.
//    MODE 0: h = bf16(relu(acc+b1)) ; MODE 1: eo = f32(acc+b2)
// -------------------------------------------------------------------------
template <int MODE>
__global__ __launch_bounds__(256)
void moe_gemm_kernel(const __bf16* __restrict__ A, const __bf16* __restrict__ Bt,
                     const float* __restrict__ bias,
                     __bf16* __restrict__ outBf, float* __restrict__ outF) {
    constexpr int TM = 128, TN = 128, TK = 32, LDK = 48;  // 96B padded rows (16B aligned)
    constexpr int TILE_ELEMS = TM * LDK;                  // 6144 bf16 per tile
    constexpr int BUF_ELEMS  = 2 * TILE_ELEMS;            // A + B per stage
    extern __shared__ __align__(16) char smem_raw[];      // 2 * BUF_ELEMS * 2B = 49152 B
    __bf16* smem = (__bf16*)smem_raw;

    int e  = blockIdx.z;
    int m0 = blockIdx.x * TM;
    int n0 = blockIdx.y * TN;
    const __bf16* Ae = A  + (size_t)e * kC * kD;
    const __bf16* Be = Bt + (size_t)e * kD * kD;
    unsigned long long saA = (unsigned long long)Ae;
    unsigned long long saB = (unsigned long long)Be;

    int tid  = threadIdx.x;
    int lane = tid & 31, wave = tid >> 5;
    int wm = wave >> 2, wn = wave & 3;
    int lhalf = lane >> 4, l16 = lane & 15;

    // async DMA of one K-stage (128x32 A tile + 128x32 B tile) into LDS buffer
    auto issue_stage = [&](int k0, int buf) {
#pragma unroll
        for (int i = 0; i < 2; ++i) {
            int li  = tid + i * 256;                 // 512 16-byte segments per tile
            int row = li >> 2, seg = li & 3;
            unsigned gA = (unsigned)((( (m0 + row) * kD ) + k0 + seg * 8) * 2);
            unsigned gB = (unsigned)((( (n0 + row) * kD ) + k0 + seg * 8) * 2);
            unsigned lA = (unsigned)((buf * BUF_ELEMS + row * LDK + seg * 8) * 2);
            unsigned lB = (unsigned)((buf * BUF_ELEMS + TILE_ELEMS + row * LDK + seg * 8) * 2);
            asm volatile("global_load_async_to_lds_b128 %0, %1, %2"
                         :: "v"(lA), "v"(gA), "s"(saA) : "memory");
            asm volatile("global_load_async_to_lds_b128 %0, %1, %2"
                         :: "v"(lB), "v"(gB), "s"(saB) : "memory");
        }
    };

    f32x8 acc[4][2] = {};

    issue_stage(0, 0);
    constexpr int NSTAGE = kD / TK;                  // 32
    for (int s = 0; s < NSTAGE; ++s) {
        int buf = s & 1;
        // my stage-s async copies have landed in LDS
        asm volatile("s_wait_asynccnt 0x0" ::: "memory");
        // all waves' copies landed AND all waves finished computing stage s-1
        __syncthreads();
        // prefetch next stage into the buffer freed by stage s-1's compute
        if (s + 1 < NSTAGE) issue_stage((s + 1) * TK, buf ^ 1);

        const __bf16* Abase = smem + buf * BUF_ELEMS;
        const __bf16* Bbase = Abase + TILE_ELEMS;

        // A frags: lane<16 -> K {0..7,16..23}; lane>=16 -> K {8..15,24..31}
        bf16x16 af[4], bfr[2];
#pragma unroll
        for (int mi = 0; mi < 4; ++mi) {
            int row = wm * 64 + mi * 16 + l16;
            bf16x8 lo = *(const bf16x8*)(Abase + row * LDK + lhalf * 8);
            bf16x8 hi = *(const bf16x8*)(Abase + row * LDK + lhalf * 8 + 16);
            af[mi] = __builtin_shufflevector(lo, hi, 0,1,2,3,4,5,6,7,8,9,10,11,12,13,14,15);
        }
        // B frags: lane = column; lane<16 -> K {0..7,16..23}; lane>=16 -> K {8..15,24..31}
#pragma unroll
        for (int ni = 0; ni < 2; ++ni) {
            int col = wn * 32 + ni * 16 + l16;
            bf16x8 lo = *(const bf16x8*)(Bbase + col * LDK + lhalf * 8);
            bf16x8 hi = *(const bf16x8*)(Bbase + col * LDK + lhalf * 8 + 16);
            bfr[ni] = __builtin_shufflevector(lo, hi, 0,1,2,3,4,5,6,7,8,9,10,11,12,13,14,15);
        }
#pragma unroll
        for (int mi = 0; mi < 4; ++mi)
#pragma unroll
            for (int ni = 0; ni < 2; ++ni)
                acc[mi][ni] = __builtin_amdgcn_wmma_f32_16x16x32_bf16(
                    false, af[mi], false, bfr[ni], (short)0, acc[mi][ni], false, false);
    }

    // epilogue: C/D layout -> VGPR r holds M=r (lanes 0-15) / M=r+8 (lanes 16-31)
#pragma unroll
    for (int mi = 0; mi < 4; ++mi) {
#pragma unroll
        for (int ni = 0; ni < 2; ++ni) {
            int col = n0 + wn * 32 + ni * 16 + l16;
            float bv = bias[(size_t)e * kD + col];
#pragma unroll
            for (int r = 0; r < 8; ++r) {
                int row = m0 + wm * 64 + mi * 16 + lhalf * 8 + r;
                float v = acc[mi][ni][r] + bv;
                if (MODE == 0) {
                    v = v > 0.f ? v : 0.f;
                    outBf[((size_t)e * kC + row) * kD + col] = (__bf16)v;
                } else {
                    outF[((size_t)e * kC + row) * kD + col] = v;
                }
            }
        }
    }
}

// -------------------------------------------------------------------------
// 6) Combine: y[t][:] = w[t] * eo[idx[t]][slot[t]][:]
// -------------------------------------------------------------------------
__global__ __launch_bounds__(256)
void combine_kernel(const float* __restrict__ eo, const int* __restrict__ idxArr,
                    const int* __restrict__ slotArr, const float* __restrict__ wcomb,
                    float* __restrict__ y) {
    int t = blockIdx.x;
    float w = wcomb[t];
    int ex = idxArr[t], s = slotArr[t];
    const float* src = eo + ((size_t)ex * kC + s) * kD;
    float* dst = y + (size_t)t * kD;
    for (int i = threadIdx.x; i < kD; i += 256) dst[i] = w * src[i];
}

// -------------------------------------------------------------------------
extern "C" void kernel_launch(void* const* d_in, const int* in_sizes, int n_in,
                              void* d_out, int out_size, void* d_ws, size_t ws_size,
                              hipStream_t stream) {
    (void)in_sizes; (void)n_in; (void)out_size; (void)ws_size;
    const float* x  = (const float*)d_in[0];
    const float* Wr = (const float*)d_in[1];
    const float* br = (const float*)d_in[2];
    const float* W1 = (const float*)d_in[3];
    const float* b1 = (const float*)d_in[4];
    const float* W2 = (const float*)d_in[5];
    const float* b2 = (const float*)d_in[6];
    float* y = (float*)d_out;

    char* p = (char*)d_ws;
    auto alloc = [&](size_t bytes) -> void* {
        void* r = (void*)p;
        p += (bytes + 255) & ~(size_t)255;
        return r;
    };
    __bf16* W1t      = (__bf16*)alloc((size_t)kE * kD * kD * sizeof(__bf16)); // 32 MB
    __bf16* W2t      = (__bf16*)alloc((size_t)kE * kD * kD * sizeof(__bf16)); // 32 MB
    __bf16* ei       = (__bf16*)alloc((size_t)kE * kC * kD * sizeof(__bf16)); // 16 MB
    __bf16* h        = (__bf16*)alloc((size_t)kE * kC * kD * sizeof(__bf16)); // 16 MB
    float*  eo       = (float*) alloc((size_t)kE * kC * kD * sizeof(float));  // 32 MB
    float*  gateArr  = (float*) alloc((size_t)kT * sizeof(float));
    float*  wcomb    = (float*) alloc((size_t)kT * sizeof(float));
    int*    idxArr   = (int*)   alloc((size_t)kT * sizeof(int));
    int*    slotArr  = (int*)   alloc((size_t)kT * sizeof(int));
    int*    slot_tok = (int*)   alloc((size_t)kE * kC * sizeof(int));

    constexpr size_t kGemmLds = 2u * 2u * 128u * 48u * sizeof(__bf16);  // 49152 B

    convert_w_kernel<<<dim3(kD / 32, kD / 32, 2 * kE), 256, 0, stream>>>(W1, W2, W1t, W2t);
    router_kernel<<<kT, 256, 0, stream>>>(x, Wr, br, idxArr, gateArr);
    assign_kernel<<<1, 256, 0, stream>>>(idxArr, gateArr, slot_tok, slotArr, wcomb);
    dispatch_kernel<<<kE * kC, 256, 0, stream>>>(x, wcomb, slot_tok, ei);
    moe_gemm_kernel<0><<<dim3(kC / 128, kD / 128, kE), 256, kGemmLds, stream>>>(ei, W1t, b1, h, nullptr);
    moe_gemm_kernel<1><<<dim3(kC / 128, kD / 128, kE), 256, kGemmLds, stream>>>(h, W2t, b2, nullptr, eo);
    combine_kernel<<<kT, 256, 0, stream>>>(eo, idxArr, slotArr, wcomb, y);
}
